// RnnModel_86397562126991
// MI455X (gfx1250) — matrix operand order repaired and checked by reading, not compiled, not run on hardware
//
#include <hip/hip_runtime.h>

#define T_STEPS 2048
#define BATCH   64
#define IN_DIM  512
#define HDIM    256
#define CHUNK   16

typedef _Float16 half8 __attribute__((ext_vector_type(8)));
typedef _Float16 v16h  __attribute__((ext_vector_type(16)));
typedef float    v8f   __attribute__((ext_vector_type(8)));

static __device__ __forceinline__ v16h cat16(half8 lo, half8 hi) {
  return __builtin_shufflevector(lo, hi, 0,1,2,3,4,5,6,7,8,9,10,11,12,13,14,15);
}

// -------- f32 -> f16 row-gather convert (also extracts batch row 63 of x) ----
__global__ void cvt_rows_f16(const float* __restrict__ src, _Float16* __restrict__ dst,
                             int cols, long rstride, long n) {
  long i = (long)blockIdx.x * blockDim.x + threadIdx.x;
  if (i >= n) return;
  long r = i / cols;
  long c = i - r * cols;
  dst[i] = (_Float16)src[r * rstride + c];
}

__global__ void zero_flags(int* __restrict__ f) {
  if (threadIdx.x < 8) f[threadIdx.x] = 0;
}

// -------- WMMA GEMM: C[MxN] = A[MxK] * W[NxK]^T + (b0+b1)[N]  (layer-0 pre) --
__global__ __launch_bounds__(128) void gemm_wmma_f16(
    const _Float16* __restrict__ A, const _Float16* __restrict__ W,
    const float* __restrict__ b0, const float* __restrict__ b1,
    float* __restrict__ C, int M, int N, int K) {
  const int lane = threadIdx.x & 31;
  const int wave = threadIdx.x >> 5;
  const int m0   = blockIdx.x << 4;
  const int n0   = wave << 6;
  const int l16  = lane & 15;
  const int hsel = lane >> 4;

  const _Float16* arow = A + (size_t)(m0 + l16) * K + (hsel << 3);
  v8f acc[4] = {v8f{}, v8f{}, v8f{}, v8f{}};

  for (int k0 = 0; k0 < K; k0 += 32) {
    v16h af = cat16(*(const half8*)(arow + k0), *(const half8*)(arow + k0 + 16));
#pragma unroll
    for (int i = 0; i < 4; ++i) {
      const _Float16* brow = W + (size_t)(n0 + (i << 4) + l16) * K + k0 + (hsel << 3);
      v16h bf = cat16(*(const half8*)(brow), *(const half8*)(brow + 16));
      acc[i] = __builtin_amdgcn_wmma_f32_16x16x32_f16(
          false, af, false, bf, (short)0, acc[i], false, false);
    }
  }

  const int mb = m0 + (hsel << 3);
#pragma unroll
  for (int i = 0; i < 4; ++i) {
    const int n = n0 + (i << 4) + l16;
    const float bias = b0[n] + b1[n];
#pragma unroll
    for (int r = 0; r < 8; ++r)
      C[(size_t)(mb + r) * N + n] = acc[i][r] + bias;
  }
}

// -------- pipelined 3-layer tanh-RNN scan: one block per layer ---------------
// Block 0: scan layer 0 from precomputed pre0.
// Blocks 1,2: per 16-step chunk, wait for producer flag, compute the chunk's
//   pre with WMMA (16 waves x one 16x16 tile, K=256 -> 8 v_wmma), stage in LDS,
//   then run 16 recurrence steps (W_hh in 128 VGPRs/thread, h via LDS).
__global__ __launch_bounds__(512) void rnn_pipeline(
    const float* __restrict__ pre0,
    const float* __restrict__ whh0, const float* __restrict__ whh1,
    const float* __restrict__ whh2,
    const _Float16* __restrict__ wih1h, const _Float16* __restrict__ wih2h,
    const float* __restrict__ bih1, const float* __restrict__ bhh1,
    const float* __restrict__ bih2, const float* __restrict__ bhh2,
    _Float16* __restrict__ hs0, _Float16* __restrict__ hs1,
    _Float16* __restrict__ hs2,
    int* __restrict__ flags, int T) {
  const int l = blockIdx.x;
  const float*    whh   = (l == 0) ? whh0 : (l == 1) ? whh1 : whh2;
  _Float16*       hsout = (l == 0) ? hs0  : (l == 1) ? hs1  : hs2;
  const _Float16* hsin  = (l == 1) ? hs0  : hs1;
  const _Float16* wih   = (l == 1) ? wih1h : wih2h;
  const float*    bA    = (l == 1) ? bih1 : bih2;
  const float*    bB    = (l == 1) ? bhh1 : bhh2;

  __shared__ float preS[CHUNK * HDIM];   // 16 KB staging for this chunk's pre
  __shared__ float hbuf[2][HDIM];

  const int tid  = threadIdx.x;
  const int row  = tid >> 1;
  const int part = tid & 1;
  const int lane = tid & 31;
  const int wave = tid >> 5;
  const int l16  = lane & 15;
  const int hsel = lane >> 4;

  // W_hh slice resident in VGPRs (32 x float4 = 128 VGPRs/thread).
  float4 w4[32];
  const float* wsrc = whh + (size_t)row * HDIM + (part << 7);
#pragma unroll
  for (int j = 0; j < 32; ++j) w4[j] = *(const float4*)(wsrc + (j << 2));

  if (tid < HDIM) hbuf[1][tid] = 0.0f;   // h(-1) = 0
  __syncthreads();

  const int nchunks = T >> 4;
  for (int c = 0; c < nchunks; ++c) {
    const int tbase = c << 4;

    if (l != 0) {
      // ---- wait for producer's chunk c, then make its stores visible ----
      if (tid == 0) {
        while (__hip_atomic_load(&flags[l - 1], __ATOMIC_RELAXED,
                                 __HIP_MEMORY_SCOPE_AGENT) <= c)
          __builtin_amdgcn_s_sleep(1);
      }
      __syncthreads();
      __threadfence();

      // ---- WMMA: preS[s][n] = hsin[tbase+s,:] . wih[n,:] + bias[n] ----
      const int n0 = wave << 4;
      const _Float16* arow = hsin + (size_t)(tbase + l16) * HDIM + (hsel << 3);
      const _Float16* brow = wih  + (size_t)(n0 + l16)    * HDIM + (hsel << 3);
      v8f acc = {};
#pragma unroll
      for (int k0 = 0; k0 < HDIM; k0 += 32) {
        v16h af = cat16(*(const half8*)(arow + k0), *(const half8*)(arow + k0 + 16));
        v16h bf = cat16(*(const half8*)(brow + k0), *(const half8*)(brow + k0 + 16));
        acc = __builtin_amdgcn_wmma_f32_16x16x32_f16(
            false, af, false, bf, (short)0, acc, false, false);
      }
      const int n = n0 + l16;
      const float bias = bA[n] + bB[n];
#pragma unroll
      for (int r = 0; r < 8; ++r)
        preS[((hsel << 3) + r) * HDIM + n] = acc[r] + bias;
      __syncthreads();
    }

    // ---- 16 sequential recurrence steps ----
    for (int s = 0; s < CHUNK; ++s) {
      const int t = tbase + s;
      const float* hv = &hbuf[(t & 1) ^ 1][part << 7];
      float p = (l == 0) ? pre0[(size_t)t * HDIM + row] : preS[s * HDIM + row];
      float s0 = 0.f, s1 = 0.f, s2 = 0.f, s3 = 0.f;
#pragma unroll
      for (int j = 0; j < 32; ++j) {
        float4 h4 = *(const float4*)(hv + (j << 2));
        s0 = fmaf(w4[j].x, h4.x, s0);
        s1 = fmaf(w4[j].y, h4.y, s1);
        s2 = fmaf(w4[j].z, h4.z, s2);
        s3 = fmaf(w4[j].w, h4.w, s3);
      }
      float sum = (s0 + s1) + (s2 + s3);
      sum += __shfl_xor(sum, 1, 32);
      if (part == 0) {
        float hn = tanhf(p + sum);
        hbuf[t & 1][row] = hn;
        hsout[(size_t)t * HDIM + row] = (_Float16)hn;
      }
      __syncthreads();
    }

    // ---- publish chunk c (release) ----
    __threadfence();
    __syncthreads();
    if (tid == 0)
      __hip_atomic_store(&flags[l], c + 1, __ATOMIC_RELEASE,
                         __HIP_MEMORY_SCOPE_AGENT);
  }
}

// -------- output head: out[t,o] = hs2[t,:] . w_out[o,:] + b_out[o] -----------
__global__ __launch_bounds__(256) void out_head(
    const _Float16* __restrict__ hs, const float* __restrict__ wout,
    const float* __restrict__ bout, float* __restrict__ out, int T) {
  int t = blockIdx.x * blockDim.x + threadIdx.x;
  if (t >= T) return;
  const _Float16* hp = hs + (size_t)t * HDIM;
  float a0 = 0.f, a1 = 0.f;
#pragma unroll 8
  for (int j = 0; j < HDIM; ++j) {
    float hv = (float)hp[j];
    a0 = fmaf(hv, wout[j],        a0);
    a1 = fmaf(hv, wout[HDIM + j], a1);
  }
  out[2 * t]     = a0 + bout[0];
  out[2 * t + 1] = a1 + bout[1];
}

extern "C" void kernel_launch(void* const* d_in, const int* in_sizes, int n_in,
                              void* d_out, int out_size, void* d_ws, size_t ws_size,
                              hipStream_t stream) {
  (void)in_sizes; (void)n_in; (void)out_size; (void)ws_size;
  const float* x    = (const float*)d_in[0];
  const float* wih0 = (const float*)d_in[1];
  const float* whh0 = (const float*)d_in[2];
  const float* bih0 = (const float*)d_in[3];
  const float* bhh0 = (const float*)d_in[4];
  const float* whh1 = (const float*)d_in[6];
  const float* bih1 = (const float*)d_in[7];
  const float* bhh1 = (const float*)d_in[8];
  const float* whh2 = (const float*)d_in[10];
  const float* bih2 = (const float*)d_in[11];
  const float* bhh2 = (const float*)d_in[12];
  const float* wih1 = (const float*)d_in[5];
  const float* wih2 = (const float*)d_in[9];
  const float* wout = (const float*)d_in[13];
  const float* bout = (const float*)d_in[14];

  char* ws = (char*)d_ws;
  _Float16* x63h  = (_Float16*)(ws + 0);        // 2 MB
  float*    pre0  = (float*)   (ws + 2097152);  // 2 MB
  _Float16* wih0h = (_Float16*)(ws + 4194304);  // 256 KB
  _Float16* wih1h = (_Float16*)(ws + 4456448);  // 128 KB
  _Float16* wih2h = (_Float16*)(ws + 4587520);  // 128 KB
  _Float16* hs0h  = (_Float16*)(ws + 4718592);  // 1 MB
  _Float16* hs1h  = (_Float16*)(ws + 5767168);  // 1 MB
  _Float16* hs2h  = (_Float16*)(ws + 6815744);  // 1 MB
  int*      flags = (int*)     (ws + 7864320);  // 32 B

  // Pack batch row 63 of x and convert weights to f16.
  {
    long n = (long)T_STEPS * IN_DIM;
    cvt_rows_f16<<<(int)((n + 255) / 256), 256, 0, stream>>>(
        x + 63 * IN_DIM, x63h, IN_DIM, (long)BATCH * IN_DIM, n);
  }
  cvt_rows_f16<<<512, 256, 0, stream>>>(wih0, wih0h, IN_DIM, IN_DIM, (long)HDIM * IN_DIM);
  cvt_rows_f16<<<256, 256, 0, stream>>>(wih1, wih1h, HDIM, HDIM, (long)HDIM * HDIM);
  cvt_rows_f16<<<256, 256, 0, stream>>>(wih2, wih2h, HDIM, HDIM, (long)HDIM * HDIM);
  zero_flags<<<1, 32, 0, stream>>>(flags);

  // Layer-0 pre: (2048 x 512) x (512 x 256)^T via WMMA.
  gemm_wmma_f16<<<T_STEPS / 16, 128, 0, stream>>>(x63h, wih0h, bih0, bhh0, pre0,
                                                  T_STEPS, HDIM, IN_DIM);

  // Pipelined 3-layer scan: 3 co-resident blocks, chunk handoff via L2 flags.
  rnn_pipeline<<<3, 512, 0, stream>>>(pre0, whh0, whh1, whh2, wih1h, wih2h,
                                      bih1, bhh1, bih2, bhh2,
                                      hs0h, hs1h, hs2h, flags, T_STEPS);

  // Head
  out_head<<<(T_STEPS + 255) / 256, 256, 0, stream>>>(hs2h, wout, bout,
                                                      (float*)d_out, T_STEPS);
}
